// MoEFFN_10411000726031
// MI455X (gfx1250) — compile-verified, matrix operand order checked by reading, not run on hardware
//
#include <hip/hip_runtime.h>
#include <hip/hip_bf16.h>
#include <math.h>
#include <stdint.h>

// ---- problem constants (match reference) ----
#define D_MODEL 1024
#define D_FF    4096
#define NEXP    4
#define NTOK    8192          // B*S = 4*2048
#define MT      32            // token tile (two 16-row M-tiles)
#define MTILES  2
#define FC      128           // F chunk per iteration
#define NWAVE   8             // 256 threads / 32

#define XS_STRIDE 1032        // halves per x-tile row (pad: 516 dw % 64 = 4)
#define HS_STRIDE 136         // halves per h-tile row  (pad:  68 dw % 64 = 4)
#define XS_HALVES (MT * XS_STRIDE)            // 33024 halves = 66048 B
#define HS_HALVES (MT * HS_STRIDE)            //  4352 halves =  8704 B
#define SMEM_BYTES (XS_HALVES * 2 + 2 * HS_HALVES * 2 + MT * NEXP * 4)  // 83968

typedef __attribute__((ext_vector_type(16))) __bf16 v16bf;
typedef __attribute__((ext_vector_type(8)))  __bf16 v8bf;
typedef __attribute__((ext_vector_type(4)))  __bf16 v4bf;
typedef __attribute__((ext_vector_type(8)))  float  v8f;

union Frag16 { v16bf v; v8bf h[2]; };

// ---------------------------------------------------------------------------
// Kernel 1: router. One wave per token: logits = x.Wr + br, softmax top-2,
// renormalized weights written as wgt[t][e] (0 for unselected experts).
// ---------------------------------------------------------------------------
__global__ void moe_router_kernel(const float* __restrict__ x,
                                  const float* __restrict__ Wr,
                                  const float* __restrict__ br,
                                  float* __restrict__ wgt) {
    const int wave = threadIdx.x >> 5;
    const int lane = threadIdx.x & 31;
    const int t = blockIdx.x * NWAVE + wave;
    if (t >= NTOK) return;

    const float* xr = x + (size_t)t * D_MODEL;
    float a0 = 0.f, a1 = 0.f, a2 = 0.f, a3 = 0.f;
    for (int i = lane; i < D_MODEL; i += 32) {
        const float xv = xr[i];
        const float* wr = Wr + i * NEXP;
        a0 += xv * wr[0]; a1 += xv * wr[1];
        a2 += xv * wr[2]; a3 += xv * wr[3];
    }
    for (int off = 16; off > 0; off >>= 1) {
        a0 += __shfl_xor(a0, off, 32);
        a1 += __shfl_xor(a1, off, 32);
        a2 += __shfl_xor(a2, off, 32);
        a3 += __shfl_xor(a3, off, 32);
    }
    if (lane == 0) {
        float l[NEXP] = { a0 + br[0], a1 + br[1], a2 + br[2], a3 + br[3] };
        int i1 = 0;
        for (int e = 1; e < NEXP; ++e) if (l[e] > l[i1]) i1 = e;
        int i2 = (i1 == 0) ? 1 : 0;
        for (int e = 0; e < NEXP; ++e) if (e != i1 && l[e] > l[i2]) i2 = e;
        const float m = fmaxf(l[i1], l[i2]);
        const float p1 = __expf(l[i1] - m);
        const float p2 = __expf(l[i2] - m);
        const float inv = 1.0f / (p1 + p2);
        float w[NEXP] = {0.f, 0.f, 0.f, 0.f};
        w[i1] = p1 * inv;
        w[i2] = p2 * inv;
        float* wo = wgt + (size_t)t * NEXP;
        wo[0] = w[0]; wo[1] = w[1]; wo[2] = w[2]; wo[3] = w[3];
    }
}

// ---------------------------------------------------------------------------
// Kernel 2: fp32 -> bf16 straight convert (x). 4 elements per thread.
// ---------------------------------------------------------------------------
__global__ void cvt_bf16_kernel(const float* __restrict__ src,
                                __bf16* __restrict__ dst, int n4) {
    const int idx = blockIdx.x * blockDim.x + threadIdx.x;
    if (idx >= n4) return;
    const float4 f = reinterpret_cast<const float4*>(src)[idx];
    v4bf o = { (__bf16)f.x, (__bf16)f.y, (__bf16)f.z, (__bf16)f.w };
    reinterpret_cast<v4bf*>(dst)[idx] = o;
}

// ---------------------------------------------------------------------------
// Kernel 3: fp32 [R][C] -> bf16 transposed [C][R], per expert (blockIdx.z).
// ---------------------------------------------------------------------------
__global__ void transpose_cvt_kernel(const float* __restrict__ src,
                                     __bf16* __restrict__ dst, int R, int C) {
    __shared__ float tile[32][33];
    const size_t eoff = (size_t)blockIdx.z * R * C;
    src += eoff; dst += eoff;
    const int xcol = blockIdx.x * 32 + threadIdx.x;
    const int y0 = blockIdx.y * 32;
    for (int j = threadIdx.y; j < 32; j += 8)
        tile[j][threadIdx.x] = src[(size_t)(y0 + j) * C + xcol];
    __syncthreads();
    const int xo = blockIdx.y * 32 + threadIdx.x;
    const int yo0 = blockIdx.x * 32;
    for (int j = threadIdx.y; j < 32; j += 8)
        dst[(size_t)(yo0 + j) * R + xo] = (__bf16)tile[threadIdx.x][j];
}

// ---------------------------------------------------------------------------
// Kernel 4: fused MoE FFN, bf16 WMMA, 32-token tiles (2 M-tiles) so every
// W1/W2 B-fragment feeds two WMMAs (halves L2 traffic vs 16-token tiles).
// x tile staged with CDNA5 async global->LDS (ASYNCcnt), no VGPR bounce.
// Routing weight folded into gelu(h): all 4 experts accumulate into one
// persistent accumulator set (16 x v8f = 128 VGPRs/lane).
// h-chunk LDS is double-buffered -> one barrier per chunk.
// ---------------------------------------------------------------------------
__global__ void __launch_bounds__(256)
moe_ffn_kernel(const __bf16* __restrict__ xb,
               const __bf16* __restrict__ W1t,
               const __bf16* __restrict__ W2t,
               const float*  __restrict__ wgt,
               const float*  __restrict__ b1,
               const float*  __restrict__ b2,
               float* __restrict__ out) {
    extern __shared__ char smem[];
    __bf16* xs = reinterpret_cast<__bf16*>(smem);                       // 32 x 1032
    __bf16* hs = reinterpret_cast<__bf16*>(smem + XS_HALVES * 2);       // 2 x (32 x 136)
    float*  wl = reinterpret_cast<float*>(smem + XS_HALVES * 2 + 2 * HS_HALVES * 2);

    const int tid   = threadIdx.x;
    const int wave  = tid >> 5;
    const int lane  = tid & 31;
    const int lhalf = lane >> 4;        // 0: lanes 0-15, 1: lanes 16-31
    const int lmod  = lane & 15;
    const int m0    = blockIdx.x * MT;  // first token of tile

    // ---- stage x tile (32 x 1024 bf16) with async global->LDS copies ----
    // LDS byte offset = low 32 bits of the generic pointer (aperture rule).
    for (int it = 0; it < 16; ++it) {
        const int lin = tid + it * 256;             // 4096 16B chunks
        const int row = lin >> 7;                   // /128
        const int c8  = lin & 127;
        const unsigned lds_off =
            (unsigned)(uintptr_t)&xs[row * XS_STRIDE + c8 * 8];
        const unsigned long long gaddr =
            (unsigned long long)(uintptr_t)&xb[(size_t)(m0 + row) * D_MODEL + c8 * 8];
        asm volatile("global_load_async_to_lds_b128 %0, %1, off"
                     :: "v"(lds_off), "v"(gaddr) : "memory");
    }
    if (tid < MT * NEXP)
        wl[tid] = wgt[(size_t)(m0 + (tid >> 2)) * NEXP + (tid & 3)];
    asm volatile("s_wait_asynccnt 0x0" ::: "memory");
    __syncthreads();

    const v8f vzero = {0.f, 0.f, 0.f, 0.f, 0.f, 0.f, 0.f, 0.f};
    v8f acc[MTILES * 8];                 // acc[mt*8 + i]
    for (int i = 0; i < MTILES * 8; ++i) acc[i] = vzero;

    int cc = 0;                          // global chunk counter (buffer parity)
    for (int e = 0; e < NEXP; ++e) {
        const __bf16* W1te = W1t + (size_t)e * D_FF * D_MODEL;
        const __bf16* W2te = W2t + (size_t)e * D_MODEL * D_FF;
        float wew[MTILES][8];
        for (int mt = 0; mt < MTILES; ++mt)
            for (int j = 0; j < 8; ++j)
                wew[mt][j] = wl[(mt * 16 + lhalf * 8 + j) * NEXP + e];

        for (int c = 0; c < D_FF / FC; ++c, ++cc) {
            __bf16* hsb = hs + (cc & 1) * HS_HALVES;

            // ---- GEMM1: two 16x16 h tiles for this wave's F-columns ----
            const int n1 = c * FC + wave * 16;
            const float bias1 = b1[e * D_FF + n1 + lmod];
            v8f h8[MTILES] = { vzero, vzero };
            for (int k = 0; k < D_MODEL; k += 32) {
                Frag16 b;
                const __bf16* bp = W1te + (size_t)(n1 + lmod) * D_MODEL + k + lhalf * 8;
                b.h[0] = *reinterpret_cast<const v8bf*>(bp);
                b.h[1] = *reinterpret_cast<const v8bf*>(bp + 16);
                #pragma unroll
                for (int mt = 0; mt < MTILES; ++mt) {
                    Frag16 a;
                    const int abase = (mt * 16 + lmod) * XS_STRIDE + k + lhalf * 8;
                    a.h[0] = *reinterpret_cast<const v8bf*>(&xs[abase]);
                    a.h[1] = *reinterpret_cast<const v8bf*>(&xs[abase + 16]);
                    h8[mt] = __builtin_amdgcn_wmma_f32_16x16x32_bf16(
                                 false, a.v, false, b.v, (short)0, h8[mt], false, false);
                }
            }
            // ---- bias + exact GELU + routing-weight fold, to LDS as bf16 ----
            #pragma unroll
            for (int mt = 0; mt < MTILES; ++mt)
                for (int j = 0; j < 8; ++j) {
                    float v = h8[mt][j] + bias1;
                    v = 0.5f * v * (1.0f + erff(v * 0.70710678118654752f));
                    v *= wew[mt][j];
                    hsb[(mt * 16 + lhalf * 8 + j) * HS_STRIDE + wave * 16 + lmod] = (__bf16)v;
                }
            __syncthreads();   // hsb complete; prior reads of this buffer are 2 chunks old
            // ---- GEMM2: out(32 x 1024) += h_chunk(32 x 128) @ W2 chunk ----
            for (int k2 = 0; k2 < FC; k2 += 32) {
                Frag16 a2[MTILES];
                #pragma unroll
                for (int mt = 0; mt < MTILES; ++mt) {
                    const int hbase = (mt * 16 + lmod) * HS_STRIDE + k2 + lhalf * 8;
                    a2[mt].h[0] = *reinterpret_cast<const v8bf*>(&hsb[hbase]);
                    a2[mt].h[1] = *reinterpret_cast<const v8bf*>(&hsb[hbase + 16]);
                }
                const int kg = c * FC + k2;
                #pragma unroll
                for (int i = 0; i < 8; ++i) {
                    const int n2 = (i * NWAVE + wave) * 16;
                    const __bf16* bp = W2te + (size_t)(n2 + lmod) * D_FF + kg + lhalf * 8;
                    Frag16 b2f;
                    b2f.h[0] = *reinterpret_cast<const v8bf*>(bp);
                    b2f.h[1] = *reinterpret_cast<const v8bf*>(bp + 16);
                    #pragma unroll
                    for (int mt = 0; mt < MTILES; ++mt)
                        acc[mt * 8 + i] = __builtin_amdgcn_wmma_f32_16x16x32_bf16(
                            false, a2[mt].v, false, b2f.v, (short)0, acc[mt * 8 + i],
                            false, false);
                }
            }
        }
    }

    // ---- epilogue: + sum_e w_e * b2_e[d], store fp32 ----
    for (int i = 0; i < 8; ++i) {
        const int d = (i * NWAVE + wave) * 16 + lmod;
        float b2v[NEXP];
        for (int e = 0; e < NEXP; ++e) b2v[e] = b2[e * D_MODEL + d];
        for (int mt = 0; mt < MTILES; ++mt)
            for (int j = 0; j < 8; ++j) {
                float bsum = 0.f;
                const int row = mt * 16 + lhalf * 8 + j;
                for (int e = 0; e < NEXP; ++e)
                    bsum += wl[row * NEXP + e] * b2v[e];
                out[(size_t)(m0 + row) * D_MODEL + d] = acc[mt * 8 + i][j] + bsum;
            }
    }
}

// ---------------------------------------------------------------------------
extern "C" void kernel_launch(void* const* d_in, const int* in_sizes, int n_in,
                              void* d_out, int out_size, void* d_ws, size_t ws_size,
                              hipStream_t stream) {
    const float* x  = (const float*)d_in[0];
    const float* Wr = (const float*)d_in[1];
    const float* br = (const float*)d_in[2];
    const float* W1 = (const float*)d_in[3];
    const float* b1 = (const float*)d_in[4];
    const float* W2 = (const float*)d_in[5];
    const float* b2 = (const float*)d_in[6];
    float* out = (float*)d_out;

    // workspace layout (bytes): xb 16MB | W1t 32MB | W2t 32MB | wgt 128KB
    char* ws = (char*)d_ws;
    __bf16* xb  = (__bf16*)(ws);
    __bf16* W1t = (__bf16*)(ws + (size_t)16777216);
    __bf16* W2t = (__bf16*)(ws + (size_t)16777216 + 33554432);
    float*  wgt = (float*) (ws + (size_t)16777216 + 33554432 + 33554432);

    moe_router_kernel<<<NTOK / NWAVE, 256, 0, stream>>>(x, Wr, br, wgt);

    {
        const int n4 = NTOK * D_MODEL / 4;
        cvt_bf16_kernel<<<(n4 + 255) / 256, 256, 0, stream>>>(x, xb, n4);
    }
    {
        dim3 g(D_FF / 32, D_MODEL / 32, NEXP), b(32, 8);
        transpose_cvt_kernel<<<g, b, 0, stream>>>(W1, W1t, D_MODEL, D_FF);
    }
    {
        dim3 g(D_MODEL / 32, D_FF / 32, NEXP), b(32, 8);
        transpose_cvt_kernel<<<g, b, 0, stream>>>(W2, W2t, D_FF, D_MODEL);
    }

    moe_ffn_kernel<<<NTOK / MT, 256, SMEM_BYTES, stream>>>(xb, W1t, W2t, wgt,
                                                           b1, b2, out);
}
